// PBoundNetwork_70274254897163
// MI455X (gfx1250) — matrix operand order repaired and checked by reading, not compiled, not run on hardware
//
#include <hip/hip_runtime.h>
#include <math.h>

#define TILE_N 128
#define KC 32
#define CB 64

typedef __attribute__((ext_vector_type(16))) __bf16 v16bf;
typedef __attribute__((ext_vector_type(8)))  float  v8f;

union FragBF { v16bf v; unsigned int u[8]; unsigned short s[16]; };
union FragF  { v8f   v; float f[8]; };

__device__ __forceinline__ unsigned short f32_to_bf16_rne(float f) {
  unsigned int u = __float_as_uint(f);
  unsigned int r = u + 0x7FFFu + ((u >> 16) & 1u);
  return (unsigned short)(r >> 16);
}
__device__ __forceinline__ float bf16_bits_to_f32(unsigned short h) {
  return __uint_as_float(((unsigned int)h) << 16);
}

// LDS flat pointer -> raw 32-bit LDS offset (flat LDS addr low 32 bits are the
// LDS byte address per ISA 10.2 aperture mapping)
#define LDS_OFF(p) ((unsigned)(size_t)(const void*)(p))

// async global->LDS copy, 16B per lane, tracked by ASYNCcnt
__device__ __forceinline__ void async_copy_b128(unsigned lds_off,
                                                const void* sbase,
                                                unsigned voff_bytes) {
  asm volatile("global_load_async_to_lds_b128 %0, %1, %2"
               :
               : "v"(lds_off), "v"(voff_bytes), "s"(sbase)
               : "memory");
}

// ---------------------------------------------------------------------------
// 0) one-time fp32 -> bf16 hi/lo conversion, row-major + transposed copies
// ---------------------------------------------------------------------------
__global__ __launch_bounds__(256) void convert_kernel(
    const float* __restrict__ x,
    unsigned short* __restrict__ xhi, unsigned short* __restrict__ xlo,
    unsigned short* __restrict__ xthi, unsigned short* __restrict__ xtlo,
    int n, int d)
{
  __shared__ unsigned short th[64][65], tl[64][65];
  int i0 = blockIdx.y * 64;   // over n
  int k0 = blockIdx.x * 64;   // over d
  int tid = threadIdx.x;
  for (int idx = tid; idx < 64 * 64; idx += 256) {
    int r = idx >> 6, c = idx & 63;
    float f = x[(size_t)(i0 + r) * d + k0 + c];
    unsigned short h = f32_to_bf16_rne(f);
    unsigned short l = f32_to_bf16_rne(f - bf16_bits_to_f32(h));
    xhi[(size_t)(i0 + r) * d + k0 + c] = h;
    xlo[(size_t)(i0 + r) * d + k0 + c] = l;
    th[r][c] = h; tl[r][c] = l;
  }
  __syncthreads();
  for (int idx = tid; idx < 64 * 64; idx += 256) {
    int r = idx >> 6, c = idx & 63;   // coalesced row r of transposed block
    xthi[(size_t)(k0 + r) * n + i0 + c] = th[c][r];
    xtlo[(size_t)(k0 + r) * n + i0 + c] = tl[c][r];
  }
}

// ---------------------------------------------------------------------------
// 1a) FAST SYRK: bf16 hi/lo split, async double-buffered LDS staging.
//     sigma = clip(x x^T / d). Triangular block grid, mirrored writes.
// ---------------------------------------------------------------------------
__global__ __launch_bounds__(256) void syrk_wmma_async_kernel(
    const unsigned short* __restrict__ xhi, const unsigned short* __restrict__ xlo,
    const unsigned short* __restrict__ xthi, const unsigned short* __restrict__ xtlo,
    float* __restrict__ sigma, int n, int d)
{
  __shared__ __attribute__((aligned(16))) unsigned short Ahi[2][TILE_N * KC];
  __shared__ __attribute__((aligned(16))) unsigned short Alo[2][TILE_N * KC];
  __shared__ __attribute__((aligned(16))) unsigned short BThi[2][KC * TILE_N];
  __shared__ __attribute__((aligned(16))) unsigned short BTlo[2][KC * TILE_N];

  int t  = blockIdx.x;
  int bi = (int)((sqrtf(8.0f * (float)t + 1.0f) - 1.0f) * 0.5f);
  while ((bi + 1) * (bi + 2) / 2 <= t) ++bi;
  while (bi * (bi + 1) / 2 > t) --bi;
  int bj = t - bi * (bi + 1) / 2;

  const int gi0 = bi * TILE_N;
  const int gj0 = bj * TILE_N;

  const int tid  = threadIdx.x;
  const int wave = tid >> 5;
  const int lane = tid & 31;
  const int wr   = wave >> 2;   // 0..1
  const int wc   = wave & 3;    // 0..3

  FragF acc[4][2];
  #pragma unroll
  for (int rt = 0; rt < 4; ++rt)
    #pragma unroll
    for (int ct = 0; ct < 2; ++ct)
      #pragma unroll
      for (int e = 0; e < 8; ++e) acc[rt][ct].f[e] = 0.0f;

  const int nkc = d / KC;

  // per-thread async staging of one chunk into buffer `buf`
  auto issue = [&](int buf, int k0) {
    #pragma unroll
    for (int tt = 0; tt < 2; ++tt) {
      // A tiles: 128 rows x 64B, 512 x 16B segments
      int sA   = tid + tt * 256;
      int rowA = sA >> 2;
      int segA = sA & 3;
      unsigned ldsA = LDS_OFF(&Ahi[buf][0]) + rowA * 64 + segA * 16;
      unsigned ldsAl = LDS_OFF(&Alo[buf][0]) + rowA * 64 + segA * 16;
      unsigned voffA = (unsigned)(((size_t)(gi0 + rowA) * d + k0) * 2) + segA * 16;
      async_copy_b128(ldsA,  xhi, voffA);
      async_copy_b128(ldsAl, xlo, voffA);
      // BT tiles (K-major): 32 rows x 256B, 512 x 16B segments
      int cB   = sA >> 4;
      int segB = sA & 15;
      unsigned ldsB  = LDS_OFF(&BThi[buf][0]) + cB * 256 + segB * 16;
      unsigned ldsBl = LDS_OFF(&BTlo[buf][0]) + cB * 256 + segB * 16;
      unsigned voffB = (unsigned)(((size_t)(k0 + cB) * n + gj0) * 2) + segB * 16;
      async_copy_b128(ldsB,  xthi, voffB);
      async_copy_b128(ldsBl, xtlo, voffB);
    }
  };

  issue(0, 0);
  for (int kc = 0; kc < nkc; ++kc) {
    const int cur = kc & 1, nxt = cur ^ 1;
    if (kc + 1 < nkc) {
      issue(nxt, (kc + 1) * KC);
      // loads complete in order: <=8 outstanding means `cur` chunk landed
      asm volatile("s_wait_asynccnt 0x8" ::: "memory");
    } else {
      asm volatile("s_wait_asynccnt 0x0" ::: "memory");
    }
    __syncthreads();

    const int M  = lane & 15;
    const int kh = lane >> 4;
    FragBF afh[4], afl[4], bfh[2], bfl[2];
    #pragma unroll
    for (int rt = 0; rt < 4; ++rt) {
      int row = wr * 64 + rt * 16 + M;
      #pragma unroll
      for (int v = 0; v < 8; ++v) {
        int kb = ((v < 4) ? 0 : 16) + 8 * kh + 2 * (v & 3);  // ISA A-layout
        afh[rt].u[v] = *(const unsigned int*)&Ahi[cur][row * KC + kb];
        afl[rt].u[v] = *(const unsigned int*)&Alo[cur][row * KC + kb];
      }
    }
    #pragma unroll
    for (int ct = 0; ct < 2; ++ct) {
      int cb = wc * 32 + ct * 16;
      #pragma unroll
      for (int v = 0; v < 8; ++v) {   // lane = N, element = K (ISA layout, same as A)
        int kb = ((v < 4) ? 0 : 16) + 8 * kh + 2 * (v & 3);
        bfh[ct].s[2 * v]     = BThi[cur][kb * TILE_N + cb + M];
        bfh[ct].s[2 * v + 1] = BThi[cur][(kb + 1) * TILE_N + cb + M];
        bfl[ct].s[2 * v]     = BTlo[cur][kb * TILE_N + cb + M];
        bfl[ct].s[2 * v + 1] = BTlo[cur][(kb + 1) * TILE_N + cb + M];
      }
    }
    #pragma unroll
    for (int rt = 0; rt < 4; ++rt)
      #pragma unroll
      for (int ct = 0; ct < 2; ++ct) {
        acc[rt][ct].v = __builtin_amdgcn_wmma_f32_16x16x32_bf16(
            false, afh[rt].v, false, bfh[ct].v, (short)0, acc[rt][ct].v, false, false);
        acc[rt][ct].v = __builtin_amdgcn_wmma_f32_16x16x32_bf16(
            false, afh[rt].v, false, bfl[ct].v, (short)0, acc[rt][ct].v, false, false);
        acc[rt][ct].v = __builtin_amdgcn_wmma_f32_16x16x32_bf16(
            false, afl[rt].v, false, bfh[ct].v, (short)0, acc[rt][ct].v, false, false);
      }
    __syncthreads();   // everyone done reading `cur` before it is re-filled
  }

  const float invd = 1.0f / (float)d;
  const int Nn = lane & 15;
  const int mh = lane >> 4;
  #pragma unroll
  for (int rt = 0; rt < 4; ++rt)
    #pragma unroll
    for (int ct = 0; ct < 2; ++ct)
      #pragma unroll
      for (int v = 0; v < 8; ++v) {   // C layout: M = v + 8*mh, N = lane&15
        int row = gi0 + wr * 64 + rt * 16 + v + 8 * mh;
        int col = gj0 + wc * 32 + ct * 16 + Nn;
        float val = fminf(1.0f, fmaxf(-1.0f, acc[rt][ct].f[v] * invd));
        sigma[(size_t)row * n + col] = val;
        if (bi != bj) sigma[(size_t)col * n + row] = val;
      }
}

// ---------------------------------------------------------------------------
// 1b) FALLBACK SYRK (small workspace): fused fp32->bf16 conversion in-loop
// ---------------------------------------------------------------------------
__global__ __launch_bounds__(256) void syrk_wmma_kernel(
    const float* __restrict__ x, float* __restrict__ sigma, int n, int d)
{
  __shared__ __attribute__((aligned(16))) unsigned short Ahi[TILE_N * KC];
  __shared__ __attribute__((aligned(16))) unsigned short Alo[TILE_N * KC];
  __shared__ __attribute__((aligned(16))) unsigned short BThi[KC * TILE_N];
  __shared__ __attribute__((aligned(16))) unsigned short BTlo[KC * TILE_N];

  int t  = blockIdx.x;
  int bi = (int)((sqrtf(8.0f * (float)t + 1.0f) - 1.0f) * 0.5f);
  while ((bi + 1) * (bi + 2) / 2 <= t) ++bi;
  while (bi * (bi + 1) / 2 > t) --bi;
  int bj = t - bi * (bi + 1) / 2;

  const int gi0 = bi * TILE_N;
  const int gj0 = bj * TILE_N;
  const int tid  = threadIdx.x;
  const int wave = tid >> 5;
  const int lane = tid & 31;
  const int wr   = wave >> 2;
  const int wc   = wave & 3;

  FragF acc[4][2];
  #pragma unroll
  for (int rt = 0; rt < 4; ++rt)
    #pragma unroll
    for (int ct = 0; ct < 2; ++ct)
      #pragma unroll
      for (int e = 0; e < 8; ++e) acc[rt][ct].f[e] = 0.0f;

  const int nkc = d / KC;
  for (int kc = 0; kc < nkc; ++kc) {
    const int k0 = kc * KC;
    __syncthreads();
    for (int idx = tid; idx < TILE_N * KC; idx += 256) {
      int r = idx >> 5;
      int c = idx & (KC - 1);
      float fa = x[(size_t)(gi0 + r) * d + (k0 + c)];
      unsigned short ha = f32_to_bf16_rne(fa);
      Ahi[r * KC + c] = ha;
      Alo[r * KC + c] = f32_to_bf16_rne(fa - bf16_bits_to_f32(ha));
      float fb = x[(size_t)(gj0 + r) * d + (k0 + c)];
      unsigned short hb = f32_to_bf16_rne(fb);
      BThi[c * TILE_N + r] = hb;
      BTlo[c * TILE_N + r] = f32_to_bf16_rne(fb - bf16_bits_to_f32(hb));
    }
    __syncthreads();

    const int M  = lane & 15;
    const int kh = lane >> 4;
    FragBF afh[4], afl[4], bfh[2], bfl[2];
    #pragma unroll
    for (int rt = 0; rt < 4; ++rt) {
      int row = wr * 64 + rt * 16 + M;
      #pragma unroll
      for (int v = 0; v < 8; ++v) {
        int kb = ((v < 4) ? 0 : 16) + 8 * kh + 2 * (v & 3);
        afh[rt].u[v] = *(const unsigned int*)&Ahi[row * KC + kb];
        afl[rt].u[v] = *(const unsigned int*)&Alo[row * KC + kb];
      }
    }
    #pragma unroll
    for (int ct = 0; ct < 2; ++ct) {
      int cb = wc * 32 + ct * 16;
      #pragma unroll
      for (int v = 0; v < 8; ++v) {
        int kb = ((v < 4) ? 0 : 16) + 8 * kh + 2 * (v & 3);
        bfh[ct].s[2 * v]     = BThi[kb * TILE_N + cb + M];
        bfh[ct].s[2 * v + 1] = BThi[(kb + 1) * TILE_N + cb + M];
        bfl[ct].s[2 * v]     = BTlo[kb * TILE_N + cb + M];
        bfl[ct].s[2 * v + 1] = BTlo[(kb + 1) * TILE_N + cb + M];
      }
    }
    #pragma unroll
    for (int rt = 0; rt < 4; ++rt)
      #pragma unroll
      for (int ct = 0; ct < 2; ++ct) {
        acc[rt][ct].v = __builtin_amdgcn_wmma_f32_16x16x32_bf16(
            false, afh[rt].v, false, bfh[ct].v, (short)0, acc[rt][ct].v, false, false);
        acc[rt][ct].v = __builtin_amdgcn_wmma_f32_16x16x32_bf16(
            false, afh[rt].v, false, bfl[ct].v, (short)0, acc[rt][ct].v, false, false);
        acc[rt][ct].v = __builtin_amdgcn_wmma_f32_16x16x32_bf16(
            false, afl[rt].v, false, bfh[ct].v, (short)0, acc[rt][ct].v, false, false);
      }
  }

  const float invd = 1.0f / (float)d;
  const int Nn = lane & 15;
  const int mh = lane >> 4;
  #pragma unroll
  for (int rt = 0; rt < 4; ++rt)
    #pragma unroll
    for (int ct = 0; ct < 2; ++ct)
      #pragma unroll
      for (int v = 0; v < 8; ++v) {
        int row = gi0 + wr * 64 + rt * 16 + v + 8 * mh;
        int col = gj0 + wc * 32 + ct * 16 + Nn;
        float val = fminf(1.0f, fmaxf(-1.0f, acc[rt][ct].f[v] * invd));
        sigma[(size_t)row * n + col] = val;
        if (bi != bj) sigma[(size_t)col * n + row] = val;
      }
}

// ---------------------------------------------------------------------------
// 2) arc-cosine kernel step (DEPTH=2 -> one step), out-of-place
// ---------------------------------------------------------------------------
__global__ void step_kernel(const float* __restrict__ sigma, float* __restrict__ out,
                            const float* __restrict__ alpha_p, int n)
{
  long long idx = (long long)blockIdx.x * blockDim.x + threadIdx.x;
  long long total = (long long)n * n;
  if (idx >= total) return;
  int i = (int)(idx / n), j = (int)(idx % n);
  float s  = sigma[idx];
  float di = sigma[(size_t)i * n + i];
  float dj = sigma[(size_t)j * n + j];
  float cross = sqrtf(di * dj);
  float m = fminf(1.0f, fmaxf(-1.0f, s / cross));
  float q = 1.0f - m * m;
  const float PI = 3.14159265358979323846f;
  float sq, ac;
  if (q > 0.0f) { sq = sqrtf(q); ac = acosf(m); }
  else          { sq = 0.0f;     ac = (m > 0.0f) ? 0.0f : PI; }
  float f  = (sq + m * (PI - ac)) / PI;
  float ts = cross * f;
  float a  = alpha_p[0], a2 = a * a;
  float r  = (s + a2 * ts) / (1.0f + a2);
  out[idx] = fminf(1.0f, fmaxf(-1.0f, r));
}

// ---------------------------------------------------------------------------
// 3) Blocked right-looking Cholesky, in place, CB=64
// ---------------------------------------------------------------------------
__global__ __launch_bounds__(256) void chol_diag_kernel(float* S, int n, int k)
{
  __shared__ float sh[CB][CB + 1];
  int tid = threadIdx.x;
  int g0 = k * CB;
  for (int idx = tid; idx < CB * CB; idx += 256) {
    int r = idx >> 6, c = idx & 63;
    sh[r][c] = S[(size_t)(g0 + r) * n + g0 + c];
  }
  __syncthreads();
  for (int j = 0; j < CB; ++j) {
    if (tid == 0) sh[j][j] = sqrtf(sh[j][j]);
    __syncthreads();
    for (int i = j + 1 + tid; i < CB; i += 256) sh[i][j] /= sh[j][j];
    __syncthreads();
    for (int idx = tid; idx < CB * CB; idx += 256) {
      int r = idx >> 6, c = idx & 63;
      if (c > j && r >= c) sh[r][c] -= sh[r][j] * sh[c][j];
    }
    __syncthreads();
  }
  for (int idx = tid; idx < CB * CB; idx += 256) {
    int r = idx >> 6, c = idx & 63;
    if (r >= c) S[(size_t)(g0 + r) * n + g0 + c] = sh[r][c];
  }
}

__global__ __launch_bounds__(64) void chol_trsm_kernel(float* S, int n, int k)
{
  __shared__ float Lkk[CB][CB + 1];
  int tid = threadIdx.x;
  int g0 = k * CB;
  int gi = g0 + CB + blockIdx.x * CB;
  for (int c = 0; c < CB; ++c) Lkk[tid][c] = S[(size_t)(g0 + tid) * n + g0 + c];
  __syncthreads();
  float xr[CB];
  for (int c = 0; c < CB; ++c) xr[c] = S[(size_t)(gi + tid) * n + g0 + c];
  for (int j = 0; j < CB; ++j) {
    float s = xr[j];
    for (int p = 0; p < j; ++p) s -= xr[p] * Lkk[j][p];
    xr[j] = s / Lkk[j][j];
  }
  for (int c = 0; c < CB; ++c) S[(size_t)(gi + tid) * n + g0 + c] = xr[c];
}

__global__ __launch_bounds__(256) void chol_update_kernel(float* S, int n, int k)
{
  __shared__ float Pi[CB][CB + 1], Pj[CB][CB + 1];
  int t  = blockIdx.x;
  int bi = (int)((sqrtf(8.0f * (float)t + 1.0f) - 1.0f) * 0.5f);
  while ((bi + 1) * (bi + 2) / 2 <= t) ++bi;
  while (bi * (bi + 1) / 2 > t) --bi;
  int bj = t - bi * (bi + 1) / 2;
  int g0 = k * CB;
  int gi = g0 + CB + bi * CB;
  int gj = g0 + CB + bj * CB;
  int tid = threadIdx.x;
  for (int idx = tid; idx < CB * CB; idx += 256) {
    int r = idx >> 6, c = idx & 63;
    Pi[r][c] = S[(size_t)(gi + r) * n + g0 + c];
    Pj[r][c] = S[(size_t)(gj + r) * n + g0 + c];
  }
  __syncthreads();
  int tr = (tid >> 4) * 4;
  int tc = (tid & 15) * 4;
  float o[4][4] = {};
  for (int p = 0; p < CB; ++p) {
    float a0 = Pi[tr + 0][p], a1 = Pi[tr + 1][p], a2 = Pi[tr + 2][p], a3 = Pi[tr + 3][p];
    float b0 = Pj[tc + 0][p], b1 = Pj[tc + 1][p], b2 = Pj[tc + 2][p], b3 = Pj[tc + 3][p];
    o[0][0] += a0 * b0; o[0][1] += a0 * b1; o[0][2] += a0 * b2; o[0][3] += a0 * b3;
    o[1][0] += a1 * b0; o[1][1] += a1 * b1; o[1][2] += a1 * b2; o[1][3] += a1 * b3;
    o[2][0] += a2 * b0; o[2][1] += a2 * b1; o[2][2] += a2 * b2; o[2][3] += a2 * b3;
    o[3][0] += a3 * b0; o[3][1] += a3 * b1; o[3][2] += a3 * b2; o[3][3] += a3 * b3;
  }
  #pragma unroll
  for (int a = 0; a < 4; ++a)
    #pragma unroll
    for (int b = 0; b < 4; ++b)
      S[(size_t)(gi + tr + a) * n + gj + tc + b] -= o[a][b];
}

// ---------------------------------------------------------------------------
// 4) trace(inv) = ||L^-1||_F^2 : per-column forward solves, 16-lane groups
// ---------------------------------------------------------------------------
__global__ __launch_bounds__(64) void trinv_trace_kernel(
    const float* __restrict__ L, float* __restrict__ trace_acc, int n)
{
  __shared__ float z[4 * 2048];
  int tid = threadIdx.x;
  int g = tid >> 4, lane = tid & 15;
  float* zc = &z[g * 2048];
  int col = blockIdx.x * 4 + g;
  float acc2 = 0.0f;
  for (int i = col; i < n; ++i) {
    float s = 0.0f;
    for (int p = col + lane; p < i; p += 16) s += L[(size_t)i * n + p] * zc[p];
    #pragma unroll
    for (int o = 8; o >= 1; o >>= 1) s += __shfl_xor(s, o, 16);
    float zi = (((i == col) ? 1.0f : 0.0f) - s) / L[(size_t)i * n + i];
    if (lane == 0) { zc[i] = zi; acc2 += zi * zi; }
  }
  if (lane == 0) atomicAdd(trace_acc, acc2);
}

// 5) y = L^-1 c (single wave); inv_proj = ||y||^2 folded into final kernel
__global__ __launch_bounds__(32) void solve_y_kernel(
    const float* __restrict__ L, const float* __restrict__ c,
    float* __restrict__ y, int n)
{
  __shared__ float ys[2048];
  int lane = threadIdx.x;
  for (int i = 0; i < n; ++i) {
    float s = 0.0f;
    for (int p = lane; p < i; p += 32) s += L[(size_t)i * n + p] * ys[p];
    #pragma unroll
    for (int o = 16; o >= 1; o >>= 1) s += __shfl_xor(s, o, 32);
    if (lane == 0) ys[i] = (c[i] - s) / L[(size_t)i * n + i];
  }
  for (int i = lane; i < n; i += 32) y[i] = ys[i];
}

__global__ void zero_kernel(float* p) { if (threadIdx.x == 0) p[0] = 0.0f; }

__global__ __launch_bounds__(256) void final_kernel(
    const float* __restrict__ L, const float* __restrict__ y,
    const float* __restrict__ trace_acc, float* __restrict__ out, int n)
{
  __shared__ float r1[256], r2[256];
  int tid = threadIdx.x;
  float ls = 0.0f, ps = 0.0f;
  for (int i = tid; i < n; i += 256) {
    ls += logf(L[(size_t)i * n + i]);
    ps += y[i] * y[i];
  }
  r1[tid] = ls; r2[tid] = ps;
  __syncthreads();
  for (int s = 128; s > 0; s >>= 1) {
    if (tid < s) { r1[tid] += r1[tid + s]; r2[tid] += r2[tid + s]; }
    __syncthreads();
  }
  if (tid == 0) {
    const float PI = 3.14159265358979323846f;
    float det_root = expf((2.0f / (float)n) * r1[0]);   // prod diag(L)^(2/n)
    out[0] = (float)n / 5.0f +
             det_root * ((0.5f - 1.0f / PI) * trace_acc[0] + r2[0] / PI);
  }
}

extern "C" void kernel_launch(void* const* d_in, const int* in_sizes, int n_in,
                              void* d_out, int out_size, void* d_ws, size_t ws_size,
                              hipStream_t stream)
{
  const float* x     = (const float*)d_in[0];
  const float* c     = (const float*)d_in[1];
  const float* alpha = (const float*)d_in[2];
  int n = in_sizes[1];
  int d = in_sizes[0] / n;

  char*  ws        = (char*)d_ws;
  size_t nn4       = (size_t)n * n * 4;
  float* sigma     = (float*)ws;                     // n*n
  float* S2        = (float*)(ws + nn4);             // n*n (becomes L)
  float* y         = (float*)(ws + 2 * nn4);         // n
  float* trace_acc = y + n;                          // 1

  size_t bf_off   = 2 * nn4 + 65536;
  size_t bf_bytes = (size_t)n * d * 2;
  unsigned short* xhi  = (unsigned short*)(ws + bf_off);
  unsigned short* xlo  = xhi + (size_t)n * d;
  unsigned short* xthi = xlo + (size_t)n * d;
  unsigned short* xtlo = xthi + (size_t)n * d;
  bool fast = (ws_size >= bf_off + 4 * bf_bytes);

  zero_kernel<<<1, 32, 0, stream>>>(trace_acc);

  int nb = n / TILE_N;
  if (fast) {
    dim3 cg(d / 64, n / 64);
    convert_kernel<<<cg, 256, 0, stream>>>(x, xhi, xlo, xthi, xtlo, n, d);
    syrk_wmma_async_kernel<<<nb * (nb + 1) / 2, 256, 0, stream>>>(
        xhi, xlo, xthi, xtlo, sigma, n, d);
  } else {
    syrk_wmma_kernel<<<nb * (nb + 1) / 2, 256, 0, stream>>>(x, sigma, n, d);
  }

  long long tot = (long long)n * n;
  step_kernel<<<(int)((tot + 255) / 256), 256, 0, stream>>>(sigma, S2, alpha, n);

  int ncb = n / CB;
  for (int k = 0; k < ncb; ++k) {
    chol_diag_kernel<<<1, 256, 0, stream>>>(S2, n, k);
    int nrem = ncb - 1 - k;
    if (nrem > 0) {
      chol_trsm_kernel<<<nrem, 64, 0, stream>>>(S2, n, k);
      chol_update_kernel<<<nrem * (nrem + 1) / 2, 256, 0, stream>>>(S2, n, k);
    }
  }

  trinv_trace_kernel<<<n / 4, 64, 0, stream>>>(S2, trace_acc, n);
  solve_y_kernel<<<1, 32, 0, stream>>>(S2, c, y, n);
  final_kernel<<<1, 256, 0, stream>>>(S2, y, trace_acc, (float*)d_out, n);
}
